// S6_86595130622191
// MI455X (gfx1250) — matrix operand order repaired and checked by reading, not compiled, not run on hardware
//
#include <hip/hip_runtime.h>
#include <math.h>

typedef __attribute__((ext_vector_type(16))) __bf16 v16bf;
typedef __attribute__((ext_vector_type(8)))  float  v8f;
typedef __attribute__((ext_vector_type(4)))  float  v4f;

#define DT_RANK 64
#define D_STATE 16
#define BATCH   2
#define SEQ     2048
#define DIN     2048
#define E_DIM   96            // DT_RANK + 2*D_STATE
#define BL      (BATCH*SEQ)   // 4096
#define NCHUNK  32
#define CHUNK_T (SEQ/NCHUNK)  // 64

// sched_barrier mask: allow SALU + VMEM (read/write) to cross; pin VALU & WMMA.
#define SCHED_PIN 0x74

__device__ __forceinline__ v16bf cvt_frag(v4f p0, v4f p1, v4f p2, v4f p3) {
  v16bf f;
#pragma unroll
  for (int i = 0; i < 4; ++i) {
    f[i]      = (__bf16)p0[i];
    f[4 + i]  = (__bf16)p1[i];
    f[8 + i]  = (__bf16)p2[i];
    f[12 + i] = (__bf16)p3[i];
  }
  return f;
}

// ---------------------------------------------------------------------------
// Kernel 1: xp[bl, 0:96] = x[bl, :] @ Wx[e, :]^T   (M=4096, N=96, K=2048)
// One wave -> one 16-row M tile x all six 16-col N tiles. sched_barriers pin
// the cvt phase and the 6-WMMA burst apart so WMMAs issue back-to-back
// (disjoint acc/B regs -> no WAR NOPs inside the burst); VMEM may still cross
// to overlap next iteration's loads with the burst.
// ---------------------------------------------------------------------------
__global__ __launch_bounds__(256) void xp_gemm_kernel(const float* __restrict__ x,
                                                      const float* __restrict__ Wx,
                                                      float* __restrict__ xp) {
  const int lane  = threadIdx.x & 31;
  const int wave  = threadIdx.x >> 5;
  const int m0    = (blockIdx.x * 8 + wave) * 16;
  const int mrow  = m0 + (lane & 15);
  const int kA    = (lane >> 4) * 8;    // A-frag K sub-offset for this half-wave
  const int kB    = (lane >> 4) * 8;    // B-frag K sub-offset

  v8f zero = {};
  v8f acc[6];
#pragma unroll
  for (int i = 0; i < 6; ++i) acc[i] = zero;

  for (int k0 = 0; k0 < DIN; k0 += 32) {
    // A fragment: 16x32 bf16; lane holds x[mrow, k0+kA+0..7] and [k0+kA+16..23]
    const float* ap = x + (size_t)mrow * DIN + k0 + kA;
    v16bf afrag = cvt_frag(*(const v4f*)(ap + 0),  *(const v4f*)(ap + 4),
                           *(const v4f*)(ap + 16), *(const v4f*)(ap + 20));

    // Load + convert ALL six B fragments first (distinct registers)
    v16bf bfrag[6];
#pragma unroll
    for (int nt = 0; nt < 6; ++nt) {
      const int ncol = nt * 16 + (lane & 15);
      const float* bp = Wx + (size_t)ncol * DIN + k0 + kB;
      bfrag[nt] = cvt_frag(*(const v4f*)(bp + 0), *(const v4f*)(bp + 4),
                           *(const v4f*)(bp + 16), *(const v4f*)(bp + 20));
    }

    __builtin_amdgcn_sched_barrier(SCHED_PIN);
#pragma unroll
    for (int nt = 0; nt < 6; ++nt)
      acc[nt] = __builtin_amdgcn_wmma_f32_16x16x32_bf16(
          false, afrag, false, bfrag[nt], (short)0, acc[nt], false, false);
    __builtin_amdgcn_sched_barrier(SCHED_PIN);
  }

  const int rbase = (lane >> 4) * 8;   // D layout: VGPR r -> M = r + 8*(lane>=16)
#pragma unroll
  for (int nt = 0; nt < 6; ++nt) {
    const int ncol = nt * 16 + (lane & 15);
#pragma unroll
    for (int r = 0; r < 8; ++r)
      xp[(size_t)(m0 + rbase + r) * E_DIM + ncol] = acc[nt][r];
  }
}

// ---------------------------------------------------------------------------
// Kernel 2: dt[bl, d] = softplus(xp[bl, 0:64] @ Wdt[d, :]^T + bdt[d])
// (M=4096, N=2048, K=64). One wave -> one 16-row M tile x eight N tiles.
// ---------------------------------------------------------------------------
__global__ __launch_bounds__(256) void dt_gemm_kernel(const float* __restrict__ xp,
                                                      const float* __restrict__ Wdt,
                                                      const float* __restrict__ bdt,
                                                      float* __restrict__ dtbuf) {
  const int lane  = threadIdx.x & 31;
  const int wave  = threadIdx.x >> 5;
  const int m0    = (blockIdx.x * 8 + wave) * 16;
  const int nbase = blockIdx.y * 128;
  const int mrow  = m0 + (lane & 15);
  const int kA    = (lane >> 4) * 8;
  const int kB    = (lane >> 4) * 8;

  v8f zero = {};
  v8f acc[8];
#pragma unroll
  for (int i = 0; i < 8; ++i) acc[i] = zero;

#pragma unroll
  for (int k0 = 0; k0 < DT_RANK; k0 += 32) {
    const float* ap = xp + (size_t)mrow * E_DIM + k0 + kA;
    v16bf afrag = cvt_frag(*(const v4f*)(ap + 0),  *(const v4f*)(ap + 4),
                           *(const v4f*)(ap + 16), *(const v4f*)(ap + 20));

    v16bf bfrag[8];
#pragma unroll
    for (int nt = 0; nt < 8; ++nt) {
      const int ncol = nbase + nt * 16 + (lane & 15);
      const float* bp = Wdt + (size_t)ncol * DT_RANK + k0 + kB;
      bfrag[nt] = cvt_frag(*(const v4f*)(bp + 0), *(const v4f*)(bp + 4),
                           *(const v4f*)(bp + 16), *(const v4f*)(bp + 20));
    }

    __builtin_amdgcn_sched_barrier(SCHED_PIN);
#pragma unroll
    for (int nt = 0; nt < 8; ++nt)
      acc[nt] = __builtin_amdgcn_wmma_f32_16x16x32_bf16(
          false, afrag, false, bfrag[nt], (short)0, acc[nt], false, false);
    __builtin_amdgcn_sched_barrier(SCHED_PIN);
  }

  const int rbase = (lane >> 4) * 8;
#pragma unroll
  for (int nt = 0; nt < 8; ++nt) {
    const int ncol = nbase + nt * 16 + (lane & 15);
    const float bv = bdt[ncol];
#pragma unroll
    for (int r = 0; r < 8; ++r) {
      float z  = acc[nt][r] + bv;
      float sp = (z > 20.0f) ? z : log1pf(__expf(z));   // softplus
      dtbuf[(size_t)(m0 + rbase + r) * DIN + ncol] = sp;
    }
  }
}

// ---------------------------------------------------------------------------
// Kernel 3 (phase 1): per-chunk local scan -> (P = prod dA, U = folded dBx)
// Thread = (b, chunk, channel). B tile staged in LDS per chunk.
// ---------------------------------------------------------------------------
__global__ __launch_bounds__(256) void scan_phase1(const float* __restrict__ x,
                                                   const float* __restrict__ xp,
                                                   const float* __restrict__ dtbuf,
                                                   const float* __restrict__ A_log,
                                                   float* __restrict__ Pbuf,
                                                   float* __restrict__ Ubuf) {
  __shared__ float Bs[CHUNK_T][D_STATE];
  const int dch   = blockIdx.x * 256 + threadIdx.x;
  const int chunk = blockIdx.y;
  const int bb    = blockIdx.z;
  const int t0    = chunk * CHUNK_T;

  for (int i = threadIdx.x; i < CHUNK_T * D_STATE; i += 256) {
    int tt = i >> 4, n = i & 15;
    Bs[tt][n] = xp[(size_t)(bb * SEQ + t0 + tt) * E_DIM + DT_RANK + n];
  }
  __syncthreads();

  float Areg[D_STATE];
#pragma unroll
  for (int n = 0; n < D_STATE; ++n)
    Areg[n] = -__expf(A_log[(size_t)dch * D_STATE + n]);

  float P[D_STATE], U[D_STATE];
#pragma unroll
  for (int n = 0; n < D_STATE; ++n) { P[n] = 1.0f; U[n] = 0.0f; }

  for (int tt = 0; tt < CHUNK_T; ++tt) {
    const size_t idx = (size_t)(bb * SEQ + t0 + tt) * DIN + dch;
    const float dt = dtbuf[idx];
    const float dx = dt * x[idx];
#pragma unroll
    for (int n = 0; n < D_STATE; ++n) {
      const float a = __expf(dt * Areg[n]);
      P[n] *= a;
      U[n]  = fmaf(a, U[n], dx * Bs[tt][n]);
    }
  }
  const size_t sbase = ((size_t)(bb * NCHUNK + chunk) * DIN + dch) * D_STATE;
#pragma unroll
  for (int n = 0; n < D_STATE; ++n) { Pbuf[sbase + n] = P[n]; Ubuf[sbase + n] = U[n]; }
}

// ---------------------------------------------------------------------------
// Kernel 4 (phase 2): sequential combine across 32 chunks.
// Thread = (b, channel, state) -> 65536 threads.
// ---------------------------------------------------------------------------
__global__ __launch_bounds__(256) void scan_phase2(const float* __restrict__ Pbuf,
                                                   const float* __restrict__ Ubuf,
                                                   float* __restrict__ Hin) {
  const int flat = blockIdx.x * 256 + threadIdx.x;   // < BATCH*DIN*D_STATE
  const int bb   = flat / (DIN * D_STATE);
  const int rem  = flat % (DIN * D_STATE);           // dch*16 + n
  float h = 0.0f;
  for (int c = 0; c < NCHUNK; ++c) {
    const size_t base = (size_t)(bb * NCHUNK + c) * (DIN * D_STATE) + rem;
    Hin[base] = h;
    h = fmaf(Pbuf[base], h, Ubuf[base]);
  }
}

// ---------------------------------------------------------------------------
// Kernel 5 (phase 3): replay each chunk from its incoming state, emit y.
// ---------------------------------------------------------------------------
__global__ __launch_bounds__(256) void scan_phase3(const float* __restrict__ x,
                                                   const float* __restrict__ xp,
                                                   const float* __restrict__ dtbuf,
                                                   const float* __restrict__ A_log,
                                                   const float* __restrict__ Dvec,
                                                   const float* __restrict__ Hin,
                                                   float* __restrict__ out) {
  __shared__ float Bs[CHUNK_T][D_STATE];
  __shared__ float Cs[CHUNK_T][D_STATE];
  const int dch   = blockIdx.x * 256 + threadIdx.x;
  const int chunk = blockIdx.y;
  const int bb    = blockIdx.z;
  const int t0    = chunk * CHUNK_T;

  for (int i = threadIdx.x; i < CHUNK_T * D_STATE; i += 256) {
    int tt = i >> 4, n = i & 15;
    const size_t row = (size_t)(bb * SEQ + t0 + tt) * E_DIM;
    Bs[tt][n] = xp[row + DT_RANK + n];
    Cs[tt][n] = xp[row + DT_RANK + D_STATE + n];
  }
  __syncthreads();

  float Areg[D_STATE];
#pragma unroll
  for (int n = 0; n < D_STATE; ++n)
    Areg[n] = -__expf(A_log[(size_t)dch * D_STATE + n]);

  float h[D_STATE];
  const size_t sbase = ((size_t)(bb * NCHUNK + chunk) * DIN + dch) * D_STATE;
#pragma unroll
  for (int n = 0; n < D_STATE; ++n) h[n] = Hin[sbase + n];

  const float Dv = Dvec[dch];

  for (int tt = 0; tt < CHUNK_T; ++tt) {
    const size_t idx = (size_t)(bb * SEQ + t0 + tt) * DIN + dch;
    const float dt = dtbuf[idx];
    const float xv = x[idx];
    const float dx = dt * xv;
    float y = 0.0f;
#pragma unroll
    for (int n = 0; n < D_STATE; ++n) {
      const float a = __expf(dt * Areg[n]);
      h[n] = fmaf(a, h[n], dx * Bs[tt][n]);
      y    = fmaf(h[n], Cs[tt][n], y);
    }
    out[idx] = fmaf(xv, Dv, y);
  }
}

// ---------------------------------------------------------------------------
extern "C" void kernel_launch(void* const* d_in, const int* in_sizes, int n_in,
                              void* d_out, int out_size, void* d_ws, size_t ws_size,
                              hipStream_t stream) {
  (void)in_sizes; (void)n_in; (void)out_size; (void)ws_size;
  const float* x     = (const float*)d_in[0];
  const float* Wx    = (const float*)d_in[1];
  const float* Wdt   = (const float*)d_in[2];
  const float* bdt   = (const float*)d_in[3];
  const float* A_log = (const float*)d_in[4];
  const float* Dvec  = (const float*)d_in[5];
  float* out = (float*)d_out;

  float* ws    = (float*)d_ws;
  float* xp    = ws;                                            // BL*96
  float* dtbuf = xp    + (size_t)BL * E_DIM;                    // BL*DIN
  float* Pbuf  = dtbuf + (size_t)BL * DIN;                      // B*NC*DIN*16
  float* Ubuf  = Pbuf  + (size_t)BATCH * NCHUNK * DIN * D_STATE;
  float* Hin   = Ubuf  + (size_t)BATCH * NCHUNK * DIN * D_STATE;

  xp_gemm_kernel<<<dim3(BL / 128), 256, 0, stream>>>(x, Wx, xp);
  dt_gemm_kernel<<<dim3(BL / 128, DIN / 128), 256, 0, stream>>>(xp, Wdt, bdt, dtbuf);
  scan_phase1<<<dim3(DIN / 256, NCHUNK, BATCH), 256, 0, stream>>>(x, xp, dtbuf, A_log,
                                                                  Pbuf, Ubuf);
  scan_phase2<<<dim3((BATCH * DIN * D_STATE) / 256), 256, 0, stream>>>(Pbuf, Ubuf, Hin);
  scan_phase3<<<dim3(DIN / 256, NCHUNK, BATCH), 256, 0, stream>>>(x, xp, dtbuf, A_log,
                                                                  Dvec, Hin, out);
}
